// RNN_21998822490627
// MI455X (gfx1250) — compile-verified, matrix operand order checked by reading, not run on hardware
//
#include <hip/hip_runtime.h>
#include <math.h>

// ---------------------------------------------------------------------------
// RNN: out[n,t] = tanh(x[n,t]@Wx + b + h_{t-1}@Wh), N=T=256, D=H=1024
// bf16 WMMA (v_wmma_f32_16x16x32_bf16), f32 accumulate.
//  - Weights pre-swizzled once into per-lane WMMA B-fragment layout (d_ws).
//  - Projection: double-buffered LDS A stage, ONE barrier per k-iter,
//    global x loads overlapped with the WMMA chain (HBM-roofline kernel).
//  - Recurrence: ONE persistent kernel, 256 steps, device-atomic grid
//    barrier between steps.
// ---------------------------------------------------------------------------

typedef __attribute__((ext_vector_type(16))) __bf16 v16bf;
typedef __attribute__((ext_vector_type(8)))  __bf16 v8bf;
typedef __attribute__((ext_vector_type(8)))  float  v8f;
typedef __attribute__((ext_vector_type(4)))  float  v4f;
typedef __attribute__((ext_vector_type(8)))  unsigned short v8us;

#define LSTRIDE 40       // LDS row stride in bf16 elements (32 data + 8 pad)
#define STEP_WGS 32      // workgroups in the persistent recurrence kernel

__device__ __forceinline__ unsigned short f32_to_bf16_rne(float f) {
  unsigned int u = __builtin_bit_cast(unsigned int, f);
  u += 0x7FFFu + ((u >> 16) & 1u);
  return (unsigned short)(u >> 16);
}

// ---------------------------------------------------------------------------
// Pre-swizzle a [1024,1024] f32 weight matrix into bf16 WMMA B-fragment order:
// dst[((kt*64 + nt)*32 + lane)*16 + i] = W[K][col], with
//   K   = kt*32 + (lane>=16 ? 8 : 0) + (i<8 ? i : i+8)
//   col = nt*16 + (lane&15)
// ---------------------------------------------------------------------------
__global__ void swizzle_weights(const float* __restrict__ W,
                                unsigned short* __restrict__ dst) {
  int t = blockIdx.x * blockDim.x + threadIdx.x;  // 0 .. 2^20-1
  int i    = t & 15;
  int lane = (t >> 4) & 31;
  int nt   = (t >> 9) & 63;
  int kt   = t >> 15;
  int K    = kt * 32 + ((lane >> 4) << 3) + (i < 8 ? i : i + 8);
  int col  = (nt << 4) + (lane & 15);
  dst[t] = f32_to_bf16_rne(W[K * 1024 + col]);
}

__global__ void init_counter(unsigned* __restrict__ c) {
  if (threadIdx.x == 0 && blockIdx.x == 0) *c = 0u;
}

// ---------------------------------------------------------------------------
// Projection: C[M,1024] = A[M,1024] @ Wx + bias.  WG = 256 thr (8 waves),
// WG tile 128x128, wave tile 16x128. Double-buffered LDS A stage, one
// barrier per k-iter, nontemporal x reads overlapped with WMMAs.
// ---------------------------------------------------------------------------
__global__ __launch_bounds__(256) void rnn_proj(
    const float* __restrict__ A,
    const unsigned short* __restrict__ Bsw,
    const float* __restrict__ bias,
    float* __restrict__ C) {
  __shared__ unsigned short As[2][128][LSTRIDE];   // ~20 KB

  const int tid  = threadIdx.x;
  const int lane = tid & 31;
  const int wave = tid >> 5;
  const long rowBase = (long)blockIdx.y * 128;
  const int  colBase = (int)blockIdx.x * 128;
  const int  ctb     = colBase >> 4;

  v8f z = {0.f, 0.f, 0.f, 0.f, 0.f, 0.f, 0.f, 0.f};
  v8f acc[8] = {z, z, z, z, z, z, z, z};

  const int arow = tid >> 1;
  const int acol = (tid & 1) << 4;
  const int m  = lane & 15;
  const int kb = (lane >> 4) << 3;

  const float* aprow = A + (rowBase + arow) * 1024 + acol;

  // prologue: fetch + convert tile k0=0 into registers
  v8us p0, p1;
  {
    const float* ap = aprow;
    v4f f0 = __builtin_nontemporal_load(reinterpret_cast<const v4f*>(ap));
    v4f f1 = __builtin_nontemporal_load(reinterpret_cast<const v4f*>(ap + 4));
    v4f f2 = __builtin_nontemporal_load(reinterpret_cast<const v4f*>(ap + 8));
    v4f f3 = __builtin_nontemporal_load(reinterpret_cast<const v4f*>(ap + 12));
#pragma unroll
    for (int j = 0; j < 4; ++j) {
      p0[j]     = f32_to_bf16_rne(f0[j]);
      p0[j + 4] = f32_to_bf16_rne(f1[j]);
      p1[j]     = f32_to_bf16_rne(f2[j]);
      p1[j + 4] = f32_to_bf16_rne(f3[j]);
    }
  }

  int buf = 0;
  for (int k0 = 0; k0 < 1024; k0 += 32) {
    // commit current tile to LDS buffer `buf`
    *reinterpret_cast<v8us*>(&As[buf][arow][acol])     = p0;
    *reinterpret_cast<v8us*>(&As[buf][arow][acol + 8]) = p1;
    __syncthreads();

    // issue next tile's global loads (overlap with WMMAs below)
    {
      const int kn = (k0 + 32 < 1024) ? (k0 + 32) : 0;  // last iter: harmless refetch
      const float* ap = aprow + kn;
      v4f f0 = __builtin_nontemporal_load(reinterpret_cast<const v4f*>(ap));
      v4f f1 = __builtin_nontemporal_load(reinterpret_cast<const v4f*>(ap + 4));
      v4f f2 = __builtin_nontemporal_load(reinterpret_cast<const v4f*>(ap + 8));
      v4f f3 = __builtin_nontemporal_load(reinterpret_cast<const v4f*>(ap + 12));
#pragma unroll
      for (int j = 0; j < 4; ++j) {
        p0[j]     = f32_to_bf16_rne(f0[j]);
        p0[j + 4] = f32_to_bf16_rne(f1[j]);
        p1[j]     = f32_to_bf16_rne(f2[j]);
        p1[j + 4] = f32_to_bf16_rne(f3[j]);
      }
    }

    // fragments + 8 WMMAs out of `buf`
    v8bf alo = *reinterpret_cast<const v8bf*>(&As[buf][wave * 16 + m][kb]);
    v8bf ahi = *reinterpret_cast<const v8bf*>(&As[buf][wave * 16 + m][kb + 16]);
    v16bf af = __builtin_shufflevector(
        alo, ahi, 0, 1, 2, 3, 4, 5, 6, 7, 8, 9, 10, 11, 12, 13, 14, 15);

    const unsigned short* bp =
        Bsw + (((size_t)(k0 >> 5) * 64 + ctb) * 32 + lane) * 16;
#pragma unroll
    for (int t = 0; t < 8; ++t) {
      v16bf bf = *reinterpret_cast<const v16bf*>(bp + t * 512);
      acc[t] = __builtin_amdgcn_wmma_f32_16x16x32_bf16(
          false, af, false, bf, (short)0, acc[t], false, false);
    }
    buf ^= 1;
  }

  const int rsub = (lane >> 4) << 3;
#pragma unroll
  for (int t = 0; t < 8; ++t) {
    const int col = colBase + t * 16 + m;
#pragma unroll
    for (int r = 0; r < 8; ++r) {
      const long row = rowBase + wave * 16 + rsub + r;
      C[row * 1024 + col] = acc[t][r] + bias[col];
    }
  }
}

// ---------------------------------------------------------------------------
// Device-scope grid barrier: monotonic counter, release-arrive / acquire-spin.
// ---------------------------------------------------------------------------
__device__ __forceinline__ void grid_barrier(unsigned* __restrict__ bar,
                                             unsigned target) {
  __threadfence();
  __syncthreads();
  if (threadIdx.x == 0) {
    __hip_atomic_fetch_add(bar, 1u, __ATOMIC_RELEASE, __HIP_MEMORY_SCOPE_AGENT);
    while (__hip_atomic_load(bar, __ATOMIC_ACQUIRE, __HIP_MEMORY_SCOPE_AGENT) <
           target) {
      __builtin_amdgcn_s_sleep(2);
    }
  }
  __syncthreads();
  __threadfence();
}

// ---------------------------------------------------------------------------
// Persistent recurrence: 256 steps of h_t = tanh(xW_t + h_{t-1} @ Wh),
// in place over out[:,t,:]. 32 WGs (8 waves), wave tile 16x64, LDS-free
// barrier-free k-loop, 4 B fragments preloaded -> back-to-back WMMAs.
// ---------------------------------------------------------------------------
__global__ __launch_bounds__(256) void rnn_recurrence(
    const float* __restrict__ h0,
    const unsigned short* __restrict__ Bsw,
    float* __restrict__ out,           // [N=256, T=256, H=1024]
    unsigned* __restrict__ bar) {
  const int tid  = threadIdx.x;
  const int lane = tid & 31;
  const int wave = tid >> 5;
  const int gwid = blockIdx.x * 8 + wave;
  const int rowTile = gwid & 15;
  const int colTile = gwid >> 4;
  const long rowBase = (long)rowTile * 16;
  const int  colBase = colTile * 64;
  const int  ctb     = colBase >> 4;
  const int m  = lane & 15;
  const int kb = (lane >> 4) << 3;
  const int rsub = (lane >> 4) << 3;
  const long ldOut = 256L * 1024L;

  const v8f z = {0.f, 0.f, 0.f, 0.f, 0.f, 0.f, 0.f, 0.f};

  for (int t = 0; t < 256; ++t) {
    const float* Aprev = (t == 0) ? h0 : (out + (size_t)(t - 1) * 1024);
    const long   ldh   = (t == 0) ? 1024L : ldOut;
    const float* ap0   = Aprev + (rowBase + m) * ldh;

    v8f acc[4] = {z, z, z, z};

#pragma unroll 4
    for (int k0 = 0; k0 < 1024; k0 += 32) {
      const float* ap = ap0 + k0 + kb;
      v4f f0 = *reinterpret_cast<const v4f*>(ap);
      v4f f1 = *reinterpret_cast<const v4f*>(ap + 4);
      v4f f2 = *reinterpret_cast<const v4f*>(ap + 16);
      v4f f3 = *reinterpret_cast<const v4f*>(ap + 20);

      const unsigned short* bp =
          Bsw + (((size_t)(k0 >> 5) * 64 + ctb) * 32 + lane) * 16;
      __builtin_prefetch(bp + (k0 + 32 < 1024 ? 32768 : 0), 0, 1);
      v16bf bfr[4];
#pragma unroll
      for (int q = 0; q < 4; ++q)
        bfr[q] = *reinterpret_cast<const v16bf*>(bp + q * 512);

      v8us lo, hi;
#pragma unroll
      for (int j = 0; j < 4; ++j) {
        lo[j]     = f32_to_bf16_rne(f0[j]);
        lo[j + 4] = f32_to_bf16_rne(f1[j]);
        hi[j]     = f32_to_bf16_rne(f2[j]);
        hi[j + 4] = f32_to_bf16_rne(f3[j]);
      }
      v16bf af = __builtin_shufflevector(
          __builtin_bit_cast(v8bf, lo), __builtin_bit_cast(v8bf, hi),
          0, 1, 2, 3, 4, 5, 6, 7, 8, 9, 10, 11, 12, 13, 14, 15);

#pragma unroll
      for (int q = 0; q < 4; ++q)
        acc[q] = __builtin_amdgcn_wmma_f32_16x16x32_bf16(
            false, af, false, bfr[q], (short)0, acc[q], false, false);
    }

    float* Ct = out + (size_t)t * 1024;
#pragma unroll
    for (int q = 0; q < 4; ++q) {
      const int col = colBase + q * 16 + m;
#pragma unroll
      for (int r = 0; r < 8; ++r) {
        const long row = rowBase + rsub + r;
        float xw = __builtin_nontemporal_load(Ct + row * ldOut + col);
        Ct[row * ldOut + col] = tanhf(acc[q][r] + xw);
      }
    }

    grid_barrier(bar, (unsigned)(t + 1) * STEP_WGS);
  }
}

extern "C" void kernel_launch(void* const* d_in, const int* in_sizes, int n_in,
                              void* d_out, int out_size, void* d_ws, size_t ws_size,
                              hipStream_t stream) {
  (void)in_sizes; (void)n_in; (void)out_size; (void)ws_size;
  const float* x  = (const float*)d_in[0];  // [N,T,D]
  const float* h0 = (const float*)d_in[1];  // [N,H]
  const float* Wx = (const float*)d_in[2];  // [D,H]
  const float* Wh = (const float*)d_in[3];  // [H,H]
  const float* b  = (const float*)d_in[4];  // [H]

  const int N = 256, T = 256;
  const int WELEMS = 1024 * 1024;

  unsigned short* WxS = (unsigned short*)d_ws;        // swizzled bf16 Wx
  unsigned short* WhS = WxS + (size_t)WELEMS;         // swizzled bf16 Wh
  unsigned* bar = (unsigned*)(WhS + (size_t)WELEMS);  // barrier counter

  swizzle_weights<<<dim3(WELEMS / 256), 256, 0, stream>>>(Wx, WxS);
  swizzle_weights<<<dim3(WELEMS / 256), 256, 0, stream>>>(Wh, WhS);
  init_counter<<<dim3(1), 32, 0, stream>>>(bar);

  float* out = (float*)d_out;  // [N,T,H]

  rnn_proj<<<dim3(1024 / 128, (N * 256) / 128), 256, 0, stream>>>(x, WxS, b, out);
  rnn_recurrence<<<dim3(STEP_WGS), 256, 0, stream>>>(h0, WhS, out, bar);
}
